// GCNConv_43396349558833
// MI455X (gfx1250) — matrix operand order repaired and checked by reading, not compiled, not run on hardware
//
#include <hip/hip_runtime.h>

typedef __attribute__((ext_vector_type(16))) _Float16 v16h;
typedef __attribute__((ext_vector_type(8)))  float    v8f;
typedef __attribute__((ext_vector_type(4)))  float    v4f;

#define D_IN  128
#define D_OUT 64

// ---------------------------------------------------------------------------
// Kernel 0: pack W [128x64] f32 -> f16 fragments in WMMA B-operand layout.
// Fragment (kt, nt) covers K = kt*32..+31, N = nt*16..+15.
// B layout used here (contiguous k, same order as A; 32x16 B, 8 VGPRs = 16 halves/lane):
//   lanes 0-15 : n = lane,    half j holds K = kt*32 + j        (j = 0..15)
//   lanes 16-31: n = lane-16, half j holds K = kt*32 + 16 + j
// Stored so the GEMM wave reads its fragment as one contiguous v16h per lane.
// ---------------------------------------------------------------------------
__global__ void gcn_prepack_w(const float* __restrict__ W,
                              _Float16* __restrict__ Bpack) {
  int tid = blockIdx.x * blockDim.x + threadIdx.x;   // 0..511
  if (tid >= 512) return;
  int lane = tid & 31;
  int frag = tid >> 5;           // 0..15
  int kt   = frag >> 2;          // K tile (4 tiles of 32)
  int nt   = frag & 3;           // N tile (4 tiles of 16)
  int n    = nt * 16 + (lane & 15);
  int kbase = kt * 32 + ((lane >> 4) << 4);   // +16 for lanes 16..31
  v16h out;
#pragma unroll
  for (int j = 0; j < 16; ++j) {
    out[j] = (_Float16)W[(kbase + j) * D_OUT + n];
  }
  *(((v16h*)Bpack) + tid) = out;
}

// ---------------------------------------------------------------------------
// Kernel 1: X' = X @ W via v_wmma_f32_16x16x32_f16.
// One wave per 16-row M tile; 4 accumulators cover N = 0..63.
// A layout used here (contiguous k, same order as B; 16x32 f16, 8 VGPRs = 16 halves/lane):
//   lanes 0-15 : m = lane,    halves 0-15 -> K 0-15
//   lanes 16-31: m = lane-16, halves 0-15 -> K 16-31
// => per lane: one contiguous run of 16 f32 from the X row (4x v4f).
// ---------------------------------------------------------------------------
__global__ void gcn_wmma_gemm(const float* __restrict__ X,
                              const _Float16* __restrict__ Bpack,
                              float* __restrict__ Xp, int N) {
  int lane = threadIdx.x & 31;
  int wave = threadIdx.x >> 5;
  int tile = blockIdx.x * (blockDim.x >> 5) + wave;
  int ntiles = (N + 15) >> 4;
  if (tile >= ntiles) return;                 // wave-uniform exit

  int m0  = tile << 4;
  int row = m0 + (lane & 15);
  if (row >= N) row = N - 1;                  // clamp (N % 16 == 0 here)
  int klo = (lane >> 4) << 4;                 // 0 or 16

  const float* xrow = X + (size_t)row * D_IN;
  const v16h*  bp   = (const v16h*)Bpack;

  v8f acc0 = {}, acc1 = {}, acc2 = {}, acc3 = {};

#pragma unroll
  for (int kt = 0; kt < 4; ++kt) {
    const float* p = xrow + kt * 32 + klo;
    v4f r0 = *(const v4f*)(p +  0);
    v4f r1 = *(const v4f*)(p +  4);
    v4f r2 = *(const v4f*)(p +  8);
    v4f r3 = *(const v4f*)(p + 12);
    v16h a;
#pragma unroll
    for (int i = 0; i < 4; ++i) {
      a[i +  0] = (_Float16)r0[i];
      a[i +  4] = (_Float16)r1[i];
      a[i +  8] = (_Float16)r2[i];
      a[i + 12] = (_Float16)r3[i];
    }
    const v16h* bb = bp + (kt * 4) * 32 + lane;
    v16h b0 = bb[0];
    v16h b1 = bb[32];
    v16h b2 = bb[64];
    v16h b3 = bb[96];
    acc0 = __builtin_amdgcn_wmma_f32_16x16x32_f16(false, a, false, b0, (short)0, acc0, false, false);
    acc1 = __builtin_amdgcn_wmma_f32_16x16x32_f16(false, a, false, b1, (short)0, acc1, false, false);
    acc2 = __builtin_amdgcn_wmma_f32_16x16x32_f16(false, a, false, b2, (short)0, acc2, false, false);
    acc3 = __builtin_amdgcn_wmma_f32_16x16x32_f16(false, a, false, b3, (short)0, acc3, false, false);
  }

  // D layout: lanes 0-15 n=lane, M=r; lanes 16-31 n=lane-16, M=r+8.
  int n   = lane & 15;
  int mhi = (lane >> 4) << 3;
#pragma unroll
  for (int r = 0; r < 8; ++r) {
    int mrow = m0 + r + mhi;
    if (mrow < N) {
      float* o = Xp + (size_t)mrow * D_OUT;
      o[n +  0] = acc0[r];
      o[n + 16] = acc1[r];
      o[n + 32] = acc2[r];
      o[n + 48] = acc3[r];
    }
  }
}

// ---------------------------------------------------------------------------
// Kernel 2: row_ptr[i] = lower_bound(row_index, i) for i in [0, N].
// ---------------------------------------------------------------------------
__global__ void gcn_rowptr(const int* __restrict__ row,
                           int* __restrict__ rp, int N, int E) {
  int i = blockIdx.x * blockDim.x + threadIdx.x;
  if (i > N) return;
  int lo = 0, hi = E;
  while (lo < hi) {
    int mid = (lo + hi) >> 1;
    if (row[mid] < i) lo = mid + 1; else hi = mid;
  }
  rp[i] = lo;
}

// ---------------------------------------------------------------------------
// Kernel 3: out[i] = sum_{e in [rp[i], rp[i+1])} Xp[col[e]].
// One wave per node; each lane owns 2 features (float2) -> coalesced 256B row
// reads that hit L2 (Xp = 12.8 MB << 192 MB L2). Unroll-by-2 for MLP.
// ---------------------------------------------------------------------------
__global__ void gcn_aggregate(const float* __restrict__ Xp,
                              const int* __restrict__ col,
                              const int* __restrict__ rp,
                              float* __restrict__ out, int N) {
  int lane = threadIdx.x & 31;
  int wave = threadIdx.x >> 5;
  int node = blockIdx.x * (blockDim.x >> 5) + wave;
  if (node >= N) return;                      // wave-uniform exit

  int e0 = rp[node];
  int e1 = rp[node + 1];

  const float2* Xp2 = (const float2*)Xp;      // row stride = 32 float2
  float2 acc; acc.x = 0.f; acc.y = 0.f;

  int e = e0;
  for (; e + 1 < e1; e += 2) {
    int c0 = col[e];
    int c1 = col[e + 1];
    float2 v0 = Xp2[(size_t)c0 * 32 + lane];
    float2 v1 = Xp2[(size_t)c1 * 32 + lane];
    acc.x += v0.x + v1.x;
    acc.y += v0.y + v1.y;
  }
  if (e < e1) {
    int c = col[e];
    float2 v = Xp2[(size_t)c * 32 + lane];
    acc.x += v.x;
    acc.y += v.y;
  }
  ((float2*)out)[(size_t)node * 32 + lane] = acc;
}

// ---------------------------------------------------------------------------
extern "C" void kernel_launch(void* const* d_in, const int* in_sizes, int n_in,
                              void* d_out, int out_size, void* d_ws, size_t ws_size,
                              hipStream_t stream) {
  const float* X   = (const float*)d_in[0];
  const float* W   = (const float*)d_in[1];
  const int*   row = (const int*)d_in[2];
  const int*   col = (const int*)d_in[3];
  float*       out = (float*)d_out;

  int N = in_sizes[0] / D_IN;   // 50000
  int E = in_sizes[2];          // 800000

  // Workspace layout: Xp [N*64 f32] | Bpack [16KB f16] | row_ptr [(N+1) i32]
  char* ws = (char*)d_ws;
  size_t xp_bytes = ((size_t)N * D_OUT * sizeof(float) + 255) & ~(size_t)255;
  float*     Xp    = (float*)ws;
  _Float16*  Bpack = (_Float16*)(ws + xp_bytes);
  size_t bp_bytes  = (512 * 16 * sizeof(_Float16) + 255) & ~(size_t)255;
  int*       rp    = (int*)(ws + xp_bytes + bp_bytes);

  gcn_prepack_w<<<2, 256, 0, stream>>>(W, Bpack);

  int ntiles = (N + 15) / 16;                    // 3125
  int waves_per_block = 8;                       // 256 threads = 8 wave32
  gcn_wmma_gemm<<<(ntiles + waves_per_block - 1) / waves_per_block,
                  32 * waves_per_block, 0, stream>>>(X, Bpack, Xp, N);

  gcn_rowptr<<<(N + 1 + 255) / 256, 256, 0, stream>>>(row, rp, N, E);

  gcn_aggregate<<<(N + waves_per_block - 1) / waves_per_block,
                  32 * waves_per_block, 0, stream>>>(Xp, col, rp, out, N);
}
